// MyLSTMAutoEncoder_83528523973187
// MI455X (gfx1250) — matrix operand order repaired and checked by reading, not compiled, not run on hardware
//
#include <hip/hip_runtime.h>
#include <hip/hip_bf16.h>
#include <cstdint>
#include <cstddef>

#define S_LEN 4096
#define EMB   768
#define HID   1536
#define G1    (4*HID)   // 6144 gates for e1/d2
#define G2    (4*EMB)   // 3072 gates for e2/d1

typedef _Float16 v8h  __attribute__((ext_vector_type(8)));
typedef _Float16 v16h __attribute__((ext_vector_type(16)));
typedef float    v8f  __attribute__((ext_vector_type(8)));

struct alignas(16) B16 { unsigned long long a, b; };

__device__ __forceinline__ v16h join16(v8h lo, v8h hi) {
  return __builtin_shufflevector(lo, hi, 0,1,2,3,4,5,6,7,8,9,10,11,12,13,14,15);
}
__device__ __forceinline__ float sigf(float x) { return 1.0f / (1.0f + __expf(-x)); }

// A fragment: lane L holds row (L%16), K = (L/16)*8 + {0..7} and {16..23}
__device__ __forceinline__ v16h ldA(const _Float16* arow, int kc) {
  const v8h* q = (const v8h*)(arow + kc);
  return join16(q[0], q[2]);
}
// B fragment: lane L holds K = (L/16)*8 + {0..7} and {16..23}
__device__ __forceinline__ v16h ldB(const _Float16* brow, int kc) {
  const v8h* q = (const v8h*)(brow + kc);
  return join16(q[0], q[2]);
}
__device__ __forceinline__ v8f wmma(v16h a, v16h b, v8f c) {
  return __builtin_amdgcn_wmma_f32_16x16x32_f16(false, a, false, b, (short)0, c,
                                                false, false);
}

// ---------------------------------------------------------------------------
// Persistent LSTM scan: grid = H/64 workgroups, 256 threads (8 waves) each.
// Each WG owns 64 hidden units -> 256 gate rows (i/f/g/o blocks of 64).
// Each wave computes 2 tiles of 16 gate rows via v_wmma_f32_16x16x32_f16 with
// B = h broadcast across all 16 columns (all D columns equal the GEMV result).
// The two tiles share one B fragment; the K loop is software-pipelined so
// each WMMA only waits on loads issued a full iteration earlier.
// Device-wide step barrier: atomic arrive + spin; h ping-pongs between hb0/hb1.
// ---------------------------------------------------------------------------
__global__ __launch_bounds__(256)
void lstm_scan_wmma(const _Float16* __restrict__ Whh,  // [4H x H] f16 row-major
                    const float* __restrict__ xW,      // [S x 4H] (single row if stride==0)
                    _Float16* __restrict__ hs_out,     // [S x H] f16 or null
                    float* __restrict__ hN_out,        // [H] or null (final h)
                    _Float16* __restrict__ hb0,
                    _Float16* __restrict__ hb1,
                    unsigned* __restrict__ sync,
                    int H, int steps, long long xw_stride)
{
  __shared__ __align__(16) _Float16 lds_h[HID];
  __shared__ float lds_g[256];

  const int tid  = (int)threadIdx.x;
  const int lane = tid & 31;
  const int wave = tid >> 5;
  const int j0   = (int)blockIdx.x * 64;
  const unsigned nwg = gridDim.x;

  // Two row-tiles per wave: tt0 = wave*2, tt1 = wave*2+1
  const int tt0 = wave * 2, tt1 = tt0 + 1;
  const int gb0 = tt0 >> 2, sb0 = tt0 & 3;
  const int gb1 = tt1 >> 2, sb1 = tt1 & 3;
  const _Float16* ar0 = Whh + ((long long)gb0 * H + j0 + sb0 * 16 + (lane & 15)) * (long long)H
                            + ((lane >> 4) << 3);
  const _Float16* ar1 = Whh + ((long long)gb1 * H + j0 + sb1 * 16 + (lane & 15)) * (long long)H
                            + ((lane >> 4) << 3);
  const _Float16* bp  = &lds_h[(lane >> 4) << 3];
  const int hv = (H * 2) / 16;  // 16-byte chunks of h

  float creg = 0.0f;  // cell state, private: one hidden unit per thread (tid<64)

  for (int t = 0; t < steps; ++t) {
    const _Float16* hbr = (t & 1) ? hb1 : hb0;
    _Float16*       hbw = (t & 1) ? hb0 : hb1;

    // Stage h(t-1) into LDS with 16-byte copies (zeroed buffers give h(-1)=0).
    for (int i = tid; i < hv; i += 256)
      ((B16*)lds_h)[i] = ((const B16*)hbr)[i];
    __syncthreads();

    // WMMA GEMV phase, software-pipelined; both tiles share the B fragment.
    v8f acc0 = {}, acc1 = {};
    v16h a0 = ldA(ar0, 0);
    v16h a1 = ldA(ar1, 0);
    v16h b  = ldB(bp, 0);
    for (int kc = 32; kc < H; kc += 32) {
      v16h na0 = ldA(ar0, kc);
      v16h na1 = ldA(ar1, kc);
      v16h nb  = ldB(bp, kc);
      __builtin_prefetch((const void*)(ar0 + kc + 32), 0, 3);
      __builtin_prefetch((const void*)(ar1 + kc + 32), 0, 3);
      acc0 = wmma(a0, b, acc0);
      acc1 = wmma(a1, b, acc1);
      a0 = na0; a1 = na1; b = nb;
    }
    acc0 = wmma(a0, b, acc0);
    acc1 = wmma(a1, b, acc1);

    // All D columns equal; lane 0 holds rows 0-7, lane 16 holds rows 8-15.
    if ((lane & 15) == 0) {
      const int half = (lane >> 4) << 3;
      const int base0 = gb0 * 64 + sb0 * 16 + half;
      const int base1 = gb1 * 64 + sb1 * 16 + half;
      #pragma unroll
      for (int r = 0; r < 8; ++r) lds_g[base0 + r] = acc0[r];
      #pragma unroll
      for (int r = 0; r < 8; ++r) lds_g[base1 + r] = acc1[r];
    }
    __syncthreads();

    // Activation phase: 64 hidden units, one per thread.
    if (tid < 64) {
      const int j = j0 + tid;
      const float* xr = xW + (long long)t * xw_stride;
      float gi = lds_g[tid]       + xr[j];
      float gf = lds_g[64 + tid]  + xr[H + j];
      float gg = lds_g[128 + tid] + xr[2 * H + j];
      float go = lds_g[192 + tid] + xr[3 * H + j];
      creg = sigf(gf) * creg + sigf(gi) * tanhf(gg);
      float h = sigf(go) * tanhf(creg);
      hbw[j] = (_Float16)h;
      if (hs_out) hs_out[(long long)t * H + j] = (_Float16)h;
      if (hN_out && t == steps - 1) hN_out[j] = h;
    }

    // Device-wide step barrier (ping-pong buffers => one barrier per step).
    __syncthreads();
    if (tid == 0) {
      __threadfence();
      atomicAdd(sync, 1u);
      const unsigned target = (unsigned)(t + 1) * nwg;
      volatile unsigned* vs = (volatile unsigned*)sync;
      while (*vs < target) __builtin_amdgcn_s_sleep(1);
    }
    __syncthreads();
    __threadfence();
  }
}

// ---------------------------------------------------------------------------
// Time-parallel input GEMM: C[t][g] = sum_k A[t][k]*W[g][k] + bih[g] + bhh[g]
// Each wave computes two 16x16 tiles (same t-rows, adjacent g-columns) sharing
// one A fragment; K loop software-pipelined.
// Grid: (S/16, G/128), block = 128 (4 waves).
// ---------------------------------------------------------------------------
__global__ __launch_bounds__(128)
void gemm_xw_wmma(const _Float16* __restrict__ A,   // [S x K] f16 row-major
                  const _Float16* __restrict__ W,   // [G x K] f16 row-major (B = W^T)
                  const float* __restrict__ bih,
                  const float* __restrict__ bhh,
                  float* __restrict__ C,            // [S x G] f32
                  int K, int G)
{
  const int lane = (int)threadIdx.x & 31;
  const int wave = (int)threadIdx.x >> 5;
  const int t0 = (int)blockIdx.x * 16;
  const int g0 = (int)blockIdx.y * 128 + wave * 32;   // tiles at g0 and g0+16

  const _Float16* ar  = A + (long long)(t0 + (lane & 15)) * K + ((lane >> 4) << 3);
  const _Float16* br0 = W + (long long)(g0 + (lane & 15)) * K + ((lane >> 4) << 3);
  const _Float16* br1 = W + (long long)(g0 + 16 + (lane & 15)) * K + ((lane >> 4) << 3);

  v8f acc0 = {}, acc1 = {};
  v16h a  = ldA(ar, 0);
  v16h b0 = ldB(br0, 0);
  v16h b1 = ldB(br1, 0);
  for (int kc = 32; kc < K; kc += 32) {
    v16h na  = ldA(ar, kc);
    v16h nb0 = ldB(br0, kc);
    v16h nb1 = ldB(br1, kc);
    __builtin_prefetch((const void*)(ar + kc + 32), 0, 3);
    __builtin_prefetch((const void*)(br0 + kc + 32), 0, 3);
    __builtin_prefetch((const void*)(br1 + kc + 32), 0, 3);
    acc0 = wmma(a, b0, acc0);
    acc1 = wmma(a, b1, acc1);
    a = na; b0 = nb0; b1 = nb1;
  }
  acc0 = wmma(a, b0, acc0);
  acc1 = wmma(a, b1, acc1);

  const int n  = lane & 15;
  const int m0 = (lane >> 4) << 3;
  const float bs0 = bih[g0 + n] + bhh[g0 + n];
  const float bs1 = bih[g0 + 16 + n] + bhh[g0 + 16 + n];
  #pragma unroll
  for (int r = 0; r < 8; ++r) {
    C[(long long)(t0 + m0 + r) * G + g0 + n]      = acc0[r] + bs0;
    C[(long long)(t0 + m0 + r) * G + g0 + 16 + n] = acc1[r] + bs1;
  }
}

// --------------------------- small helper kernels ---------------------------
__global__ void zero_u32(unsigned* __restrict__ p, long long n) {
  long long i = (long long)blockIdx.x * blockDim.x + threadIdx.x;
  if (i < n) p[i] = 0u;
}

__global__ void cvt_f32_f16(_Float16* __restrict__ dst, const float* __restrict__ src,
                            long long n) {
  long long i = (long long)blockIdx.x * blockDim.x + threadIdx.x;
  if (i < n) dst[i] = (_Float16)src[i];
}

// e1 input contribution: in_dim == 1, so xW1[t][g] = x[t]*Wih[g] + bih[g]+bhh[g]
__global__ void prep_e1(float* __restrict__ xW, const float* __restrict__ x,
                        const float* __restrict__ Wih, const float* __restrict__ bih,
                        const float* __restrict__ bhh) {
  long long i = (long long)blockIdx.x * blockDim.x + threadIdx.x;
  if (i < (long long)S_LEN * G1) {
    int t = (int)(i / G1);
    int g = (int)(i % G1);
    xW[i] = x[t] * Wih[g] + bih[g] + bhh[g];
  }
}

// d1 input contribution: input is constant z, one GEMV row of size G2
__global__ void gemv_d1(float* __restrict__ xW3, const float* __restrict__ z,
                        const float* __restrict__ W, const float* __restrict__ bih,
                        const float* __restrict__ bhh) {
  int g = (int)(blockIdx.x * blockDim.x + threadIdx.x);
  if (g < G2) {
    float s = bih[g] + bhh[g];
    const float* wr = W + (long long)g * EMB;
    for (int k = 0; k < EMB; ++k) s += z[k] * wr[k];
    xW3[g] = s;
  }
}

// Final projection: out[t] = dot(hs4[t], out_W) + out_b
__global__ __launch_bounds__(256)
void out_proj(float* __restrict__ out, const _Float16* __restrict__ hs4,
              const float* __restrict__ outW, const float* __restrict__ outB) {
  __shared__ float red[256];
  const int t = (int)blockIdx.x;
  const int tid = (int)threadIdx.x;
  float s = 0.0f;
  for (int j = tid; j < HID; j += 256)
    s += (float)hs4[(long long)t * HID + j] * outW[j];
  red[tid] = s;
  __syncthreads();
  for (int off = 128; off > 0; off >>= 1) {
    if (tid < off) red[tid] += red[tid + off];
    __syncthreads();
  }
  if (tid == 0) out[t] = red[0] + outB[0];
}

// ---------------------------------------------------------------------------
extern "C" void kernel_launch(void* const* d_in, const int* in_sizes, int n_in,
                              void* d_out, int out_size, void* d_ws, size_t ws_size,
                              hipStream_t stream)
{
  (void)in_sizes; (void)n_in; (void)out_size; (void)ws_size;

  const float* x      = (const float*)d_in[0];
  const float* e1_Wih = (const float*)d_in[1];
  const float* e1_Whh = (const float*)d_in[2];
  const float* e1_bih = (const float*)d_in[3];
  const float* e1_bhh = (const float*)d_in[4];
  const float* e2_Wih = (const float*)d_in[5];
  const float* e2_Whh = (const float*)d_in[6];
  const float* e2_bih = (const float*)d_in[7];
  const float* e2_bhh = (const float*)d_in[8];
  const float* d1_Wih = (const float*)d_in[9];
  const float* d1_Whh = (const float*)d_in[10];
  const float* d1_bih = (const float*)d_in[11];
  const float* d1_bhh = (const float*)d_in[12];
  const float* d2_Wih = (const float*)d_in[13];
  const float* d2_Whh = (const float*)d_in[14];
  const float* d2_bih = (const float*)d_in[15];
  const float* d2_bhh = (const float*)d_in[16];
  const float* out_W  = (const float*)d_in[17];
  const float* out_b  = (const float*)d_in[18];
  float* out = (float*)d_out;

  // Bump allocator over the workspace (256 B aligned).
  char* ws = (char*)d_ws;
  size_t off = 0;
  auto alloc = [&](size_t bytes) -> void* {
    void* p = ws + off;
    off = (off + bytes + 255) & ~(size_t)255;
    return p;
  };

  _Float16* wh1h = (_Float16*)alloc((size_t)G1 * HID * 2);  // e1_Whh f16
  _Float16* wh2h = (_Float16*)alloc((size_t)G2 * EMB * 2);  // e2_Whh f16
  _Float16* wh3h = (_Float16*)alloc((size_t)G2 * EMB * 2);  // d1_Whh f16
  _Float16* wh4h = (_Float16*)alloc((size_t)G1 * HID * 2);  // d2_Whh f16
  _Float16* wi2h = (_Float16*)alloc((size_t)G2 * HID * 2);  // e2_Wih f16
  _Float16* wi4h = (_Float16*)alloc((size_t)G1 * EMB * 2);  // d2_Wih f16
  float*    xw1  = (float*)   alloc((size_t)S_LEN * G1 * 4); // reused as xw4
  float*    xw2  = (float*)   alloc((size_t)S_LEN * G2 * 4);
  float*    xw3  = (float*)   alloc((size_t)G2 * 4);
  _Float16* hs1  = (_Float16*)alloc((size_t)S_LEN * HID * 2); // reused as hs4
  _Float16* hs3  = (_Float16*)alloc((size_t)S_LEN * EMB * 2);
  float*    zvec = (float*)   alloc((size_t)EMB * 4);
  unsigned* state = (unsigned*)alloc(256 + 2 * (size_t)HID * 2); // sync + hb0 + hb1
  unsigned* sync = state;
  _Float16* hb0 = (_Float16*)((char*)state + 256);
  _Float16* hb1 = hb0 + HID;
  const long long stateWords = (256 + 2 * (long long)HID * 2) / 4;

  auto cvt = [&](_Float16* dst, const float* src, long long n) {
    cvt_f32_f16<<<(unsigned)((n + 255) / 256), 256, 0, stream>>>(dst, src, n);
  };
  auto zero_state = [&]() {
    zero_u32<<<(unsigned)((stateWords + 255) / 256), 256, 0, stream>>>(state, stateWords);
  };

  // Weight precision conversion (fp32 -> fp16; safe: |w| < 0.03, fp32 accum).
  cvt(wh1h, e1_Whh, (long long)G1 * HID);
  cvt(wh2h, e2_Whh, (long long)G2 * EMB);
  cvt(wh3h, d1_Whh, (long long)G2 * EMB);
  cvt(wh4h, d2_Whh, (long long)G1 * HID);
  cvt(wi2h, e2_Wih, (long long)G2 * HID);
  cvt(wi4h, d2_Wih, (long long)G1 * EMB);

  // e1 input contribution (outer product, in_dim == 1).
  {
    long long n = (long long)S_LEN * G1;
    prep_e1<<<(unsigned)((n + 255) / 256), 256, 0, stream>>>(xw1, x, e1_Wih, e1_bih, e1_bhh);
  }

  // e1 scan: 24 WGs, hs1 = full hidden sequence (f16).
  zero_state();
  lstm_scan_wmma<<<HID / 64, 256, 0, stream>>>(wh1h, xw1, hs1, nullptr,
                                               hb0, hb1, sync, HID, S_LEN, (long long)G1);

  // e2 input GEMM: [4096 x 1536] @ [1536 x 3072]  (WMMA f16)
  gemm_xw_wmma<<<dim3(S_LEN / 16, G2 / 128), 128, 0, stream>>>(hs1, wi2h, e2_bih, e2_bhh,
                                                               xw2, HID, G2);

  // e2 scan: only final hidden state z is needed.
  zero_state();
  lstm_scan_wmma<<<EMB / 64, 256, 0, stream>>>(wh2h, xw2, nullptr, zvec,
                                               hb0, hb1, sync, EMB, S_LEN, (long long)G2);

  // d1 input contribution: constant z -> single GEMV row (stride 0 in scan).
  gemv_d1<<<(G2 + 255) / 256, 256, 0, stream>>>(xw3, zvec, d1_Wih, d1_bih, d1_bhh);

  // d1 scan.
  zero_state();
  lstm_scan_wmma<<<EMB / 64, 256, 0, stream>>>(wh3h, xw3, hs3, nullptr,
                                               hb0, hb1, sync, EMB, S_LEN, 0LL);

  // d2 input GEMM: [4096 x 768] @ [768 x 6144]  (WMMA f16; xw4 aliases xw1)
  gemm_xw_wmma<<<dim3(S_LEN / 16, G1 / 128), 128, 0, stream>>>(hs3, wi4h, d2_bih, d2_bhh,
                                                               xw1, EMB, G1);

  // d2 scan (hs4 aliases hs1).
  zero_state();
  lstm_scan_wmma<<<HID / 64, 256, 0, stream>>>(wh4h, xw1, hs1, nullptr,
                                               hb0, hb1, sync, HID, S_LEN, (long long)G1);

  // Final projection 1536 -> 1.
  out_proj<<<S_LEN, 256, 0, stream>>>(out, hs1, out_W, out_b);
}
